// AlignBlock_34239479284401
// MI455X (gfx1250) — matrix operand order repaired and checked by reading, not compiled, not run on hardware
//
#include <hip/hip_runtime.h>

// ---------------- CDNA5 WMMA types (wave32) ----------------
typedef __attribute__((ext_vector_type(16))) __bf16 v16bf;
typedef __attribute__((ext_vector_type(2)))  __bf16 v2bf;
typedef __attribute__((ext_vector_type(2)))  float  v2f;
typedef __attribute__((ext_vector_type(8)))  float  v8f;

union BfTile { v16bf v; unsigned u[8]; };

// fp32 -> bf16 via compiler fptrunc: backend selects the native (packed)
// convert op on gfx1250 if present, else legalizes to correct RNE code.
__device__ __forceinline__ unsigned pack2bf(float f0, float f1) {
  v2f in; in.x = f0; in.y = f1;
  union { v2bf v; unsigned u; } r;
  r.v = __builtin_convertvector(in, v2bf);
  return r.u;
}
__device__ __forceinline__ unsigned short f2bf(float f) {
  union { __bf16 b; unsigned short s; } r;
  r.b = (__bf16)f;
  return r.s;
}

static constexpr int Bn = 2, Cn = 64, Hn = 64, Tn = 512, Fn = 64, Dn = 32;
static constexpr int TF = Tn * Fn; // 32768

// =====================================================================
// Kernel 1: 1x1-conv projection  out[h, col] = sum_c W[h,c] * X[c,col] + bias[h]
// GEMM M=H=64, K=C=64, N=T*F per batch. Weights staged in LDS as bf16.
// Each wave: 4 m-tiles x one 16-col strip, K in 2 steps of 32 (bf16 WMMA).
// =====================================================================
__global__ __launch_bounds__(256)
void proj_kernel(const float* __restrict__ x, const float* __restrict__ w,
                 const float* __restrict__ bias, unsigned short* __restrict__ obf) {
  __shared__ unsigned short wlds[Hn * Cn];  // bf16 weights, row-major [h][c]
  __shared__ float blds[Hn];
  int tid = threadIdx.x;
  for (int i = tid; i < Hn * Cn; i += 256) wlds[i] = f2bf(w[i]);
  if (tid < Hn) blds[tid] = bias[tid];
  __syncthreads();

  int lane = tid & 31, wave = tid >> 5;
  int n = lane & 15, hl = lane >> 4;           // lane column, lane half
  int b = blockIdx.x >> 8;                     // 256 blocks per batch
  int colblk = blockIdx.x & 255;
  int col = colblk * 128 + wave * 16 + n;      // flat (t,f) column

  const float* xb = x + (size_t)b * Cn * TF;

  v8f acc[4] = {};
  for (int ks = 0; ks < 2; ++ks) {
    // B operand: 32(c) x 16(col). Layout: lanes0-15 K=0..15, lanes16-31 K=16..31,
    // VGPR s holds pair (2s, 2s+1). Rows of X are TF apart -> gather + pack bf16.
    BfTile Bt;
    #pragma unroll
    for (int s = 0; s < 8; ++s) {
      int c0 = ks * 32 + hl * 16 + 2 * s;
      float f0 = xb[(size_t)c0 * TF + col];
      float f1 = xb[(size_t)(c0 + 1) * TF + col];
      Bt.u[s] = pack2bf(f0, f1);
    }
    // A operand: 16(h) x 32(c) per m-tile, same contiguous k order as B.
    #pragma unroll
    for (int mt = 0; mt < 4; ++mt) {
      BfTile At;
      int m = mt * 16 + n;
      #pragma unroll
      for (int s = 0; s < 8; ++s) {
        int k = hl * 16 + 2 * s;
        At.u[s] = *(const unsigned*)&wlds[m * Cn + ks * 32 + k];
      }
      acc[mt] = __builtin_amdgcn_wmma_f32_16x16x32_bf16(
          false, At.v, false, Bt.v, (short)0, acc[mt], false, false);
    }
  }
  // C/D layout: VGPR r -> M=r (lanes0-15) / M=r+8 (lanes16-31)
  unsigned short* ob = obf + (size_t)b * Hn * TF;
  #pragma unroll
  for (int mt = 0; mt < 4; ++mt)
    #pragma unroll
    for (int r = 0; r < 8; ++r) {
      int h = mt * 16 + r + 8 * hl;
      ob[(size_t)h * TF + col] = f2bf(acc[mt][r] + blds[h]);
    }
}

// =====================================================================
// Kernel 2: banded Gram  V[bh,t,d] = (1/8) * sum_f Q[t,f] K[t+d-31,f]
// Per (bh, 16-row t-tile): 16x48 band of Q.K^T as 3 WMMA n-tiles, K=F=64.
// Q/K are bf16 row-major along f -> operand dword pairs load directly.
// =====================================================================
__global__ __launch_bounds__(256)
void band_kernel(const unsigned short* __restrict__ Qbf,
                 const unsigned short* __restrict__ Kbf,
                 float* __restrict__ V) {
  int tid = threadIdx.x;
  int lane = tid & 31, wave = tid >> 5;
  int n = lane & 15, hl = lane >> 4;
  int task = blockIdx.x * 8 + wave;    // 4096 tasks = 128 (b*h) x 32 t-tiles
  int tt = task & 31, bh = task >> 5;
  int t0 = tt * 16;
  const unsigned short* Qp = Qbf + (size_t)bh * TF;
  const unsigned short* Kp = Kbf + (size_t)bh * TF;

  v8f acc[3] = {};
  for (int ks = 0; ks < 2; ++ks) {
    BfTile At;                           // A = Q rows t0..t0+15, k=f
    int m = t0 + n;
    #pragma unroll
    for (int s = 0; s < 8; ++s) {
      int k = hl * 16 + 2 * s;
      At.u[s] = *(const unsigned*)&Qp[(size_t)m * Fn + ks * 32 + k];
    }
    #pragma unroll
    for (int jt = 0; jt < 3; ++jt) {     // j-tiles at t0-32, t0-16, t0
      int j = t0 - 32 + jt * 16 + n;     // B column = K frame index
      BfTile Bt;                         // B[f][j] = K[j,f]  (K^T)
      #pragma unroll
      for (int s = 0; s < 8; ++s) {
        int f = ks * 32 + hl * 16 + 2 * s;
        Bt.u[s] = (j >= 0) ? *(const unsigned*)&Kp[(size_t)j * Fn + f] : 0u;
      }
      acc[jt] = __builtin_amdgcn_wmma_f32_16x16x32_bf16(
          false, At.v, false, Bt.v, (short)0, acc[jt], false, false);
    }
  }
  float* Vp = V + (size_t)bh * Tn * Dn;
  #pragma unroll
  for (int jt = 0; jt < 3; ++jt) {
    int j0 = t0 - 32 + jt * 16;
    #pragma unroll
    for (int r = 0; r < 8; ++r) {
      int t = t0 + r + 8 * hl;
      int d = (j0 + n) - t + 31;                      // band -> window offset
      if (d >= 0 && d < Dn) Vp[(size_t)t * Dn + d] = acc[jt][r] * 0.125f;
    }
  }
}

// =====================================================================
// Kernel 3: (5,3) conv H->1 over (T,d), causal T pad 4 / d pad 1, + softmax(d).
// One wave per (b,t); lane = d (wave32 == DMAX). Shuffle-tree reductions.
// =====================================================================
__global__ __launch_bounds__(256)
void conv_softmax_kernel(const float* __restrict__ V, const float* __restrict__ wconv,
                         const float* __restrict__ bconv, float* __restrict__ A) {
  __shared__ float wc[Hn * 15];
  int tid = threadIdx.x;
  for (int i = tid; i < Hn * 15; i += 256) wc[i] = wconv[i];
  __syncthreads();
  int lane = tid & 31, wave = tid >> 5;
  int task = blockIdx.x * 8 + wave;           // b*T + t
  int b = task >> 9, t = task & (Tn - 1);
  int d = lane;
  float s = bconv[0];
  for (int h = 0; h < Hn; ++h) {
    const float* Vh = V + (size_t)(b * Hn + h) * Tn * Dn;
    const float* w3 = &wc[h * 15];
    #pragma unroll
    for (int kt = 0; kt < 5; ++kt) {
      int ts = t - 4 + kt;
      if (ts < 0) continue;                    // causal zero pad in T
      const float* r = Vh + (size_t)ts * Dn;
      float v0  = r[d];
      float vm1 = (d > 0)  ? r[d - 1] : 0.f;   // d pad
      float vp1 = (d < 31) ? r[d + 1] : 0.f;
      s = fmaf(w3[kt * 3 + 0], vm1, s);
      s = fmaf(w3[kt * 3 + 1], v0,  s);
      s = fmaf(w3[kt * 3 + 2], vp1, s);
    }
  }
  float mx = s;
  #pragma unroll
  for (int o = 16; o > 0; o >>= 1) mx = fmaxf(mx, __shfl_xor(mx, o, 32));
  float e = __expf(s - mx);
  float sum = e;
  #pragma unroll
  for (int o = 16; o > 0; o >>= 1) sum += __shfl_xor(sum, o, 32);
  A[(size_t)task * Dn + d] = e / sum;
}

// =====================================================================
// Kernel 4: aligned[b,c,t,f] = sum_d A[b,t,d] * x_ref[b,c,t+d-31,f]
// One wave per (b,c,t); float2 per lane covers F=64; A broadcast via shuffle.
// =====================================================================
__global__ __launch_bounds__(256)
void align_kernel(const float* __restrict__ xref, const float* __restrict__ A,
                  float* __restrict__ out) {
  int tid = threadIdx.x;
  int lane = tid & 31, wave = tid >> 5;
  int task = blockIdx.x * 8 + wave;          // bc*T + t
  int t = task & (Tn - 1);
  int bc = task >> 9;                        // b*C + c
  int b = bc >> 6;
  float a = A[((size_t)b * Tn + t) * Dn + lane];   // lane d holds A[b,t,d]
  const float* xr = xref + (size_t)bc * TF;
  float ax = 0.f, ay = 0.f;
  for (int dd = 0; dd < Dn; ++dd) {
    int j = t + dd - 31;
    if (j < 0) continue;                     // zero-padded frames
    if (j + 1 < Tn) __builtin_prefetch(xr + (size_t)(j + 1) * Fn + 2 * lane, 0, 0);
    float av = __shfl(a, dd, 32);
    const float2* row = (const float2*)(xr + (size_t)j * Fn);
    float2 v = row[lane];
    ax = fmaf(av, v.x, ax);
    ay = fmaf(av, v.y, ay);
  }
  float2 res; res.x = ax; res.y = ay;
  ((float2*)(out + ((size_t)bc * Tn + t) * Fn))[lane] = res;
}

// =====================================================================
extern "C" void kernel_launch(void* const* d_in, const int* in_sizes, int n_in,
                              void* d_out, int out_size, void* d_ws, size_t ws_size,
                              hipStream_t stream) {
  (void)in_sizes; (void)n_in; (void)out_size; (void)ws_size;
  const float* x_mic  = (const float*)d_in[0];
  const float* x_ref  = (const float*)d_in[1];
  const float* w_mic  = (const float*)d_in[2];
  const float* b_mic  = (const float*)d_in[3];
  const float* w_ref  = (const float*)d_in[4];
  const float* b_ref  = (const float*)d_in[5];
  const float* w_conv = (const float*)d_in[6];
  const float* b_conv = (const float*)d_in[7];
  float* out = (float*)d_out;

  // Workspace layout (all L2-resident, ~25.3 MB total):
  //   Qbf, Kbf : (B,H,T,F) bf16   8 MB each
  //   V        : (B,H,T,Dn) f32   8 MB
  //   A        : (B,T,Dn)  f32    128 KB
  char* ws = (char*)d_ws;
  size_t qkElems = (size_t)Bn * Hn * Tn * Fn;                      // 4,194,304
  unsigned short* Qbf = (unsigned short*)ws;
  unsigned short* Kbf = (unsigned short*)(ws + qkElems * 2);
  float* V = (float*)(ws + qkElems * 4);
  float* A = (float*)(ws + qkElems * 4 + (size_t)Bn * Hn * Tn * Dn * 4);

  proj_kernel<<<dim3(Bn * 256), dim3(256), 0, stream>>>(x_mic, w_mic, b_mic, Qbf);
  proj_kernel<<<dim3(Bn * 256), dim3(256), 0, stream>>>(x_ref, w_ref, b_ref, Kbf);
  band_kernel<<<dim3(Bn * Hn * 32 / 8), dim3(256), 0, stream>>>(Qbf, Kbf, V);
  conv_softmax_kernel<<<dim3(Bn * Tn / 8), dim3(256), 0, stream>>>(V, w_conv, b_conv, A);
  align_kernel<<<dim3(Bn * Cn * Tn / 8), dim3(256), 0, stream>>>(x_ref, A, out);
}
